// ExpertFeedForward_74560632258831
// MI455X (gfx1250) — matrix operand order repaired and checked by reading, not compiled, not run on hardware
//
#include <hip/hip_runtime.h>

// ---------------------------------------------------------------------------
// MoE SwiGLU feed-forward for MI455X (gfx1250, wave32, WMMA), top-2 routed.
//
// Roofline: 402 MB fp32 expert weights stream once from HBM -> ~17us @ 23.3
// TB/s. Routed top-2 compute (0.41 TFLOP bf16) sits below that -> memory
// bound. Weights convert f32->bf16 in-register during LDS staging; x (16MB
// bf16) and compacted per-expert H (bf16) stay L2-resident (192MB).
// Activation tiles (already bf16) use async global->LDS DMA (ASYNCcnt path),
// overlapped with the weight-conversion VALU work.
// ---------------------------------------------------------------------------

typedef __bf16 bf16;
typedef __attribute__((ext_vector_type(16))) __bf16 v16bf;
typedef __attribute__((ext_vector_type(8)))  __bf16 v8bf;
typedef __attribute__((ext_vector_type(8)))  float  v8f;
typedef int v4i_vs __attribute__((vector_size(16)));   // matches builtin param type

#if __has_builtin(__builtin_amdgcn_global_load_async_to_lds_b128)
#define HAVE_ASYNC 1
#else
#define HAVE_ASYNC 0
#endif
#if __has_builtin(__builtin_amdgcn_s_wait_asynccnt)
#define HAVE_WAITASYNC 1
#else
#define HAVE_WAITASYNC 0
#endif

#define TOKENS 8192
#define DMODEL 1024
#define DFF    4096
#define NEXP   8

#define BM 128          // block tile M (compacted tokens)
#define BN 64           // block tile N
#define BK 32           // K step (one bf16 WMMA K)
#define LDSTR 40        // padded LDS row stride in bf16 elems (80B, conflict-free)

// global(bf16) -> LDS 16-byte copy: async DMA if available, else via VGPRs
static __device__ __forceinline__ void copy16_g2l(const bf16* __restrict__ g,
                                                  bf16* __restrict__ l) {
#if HAVE_ASYNC
    __builtin_amdgcn_global_load_async_to_lds_b128(
        (__attribute__((address_space(1))) v4i_vs*)(g),
        (__attribute__((address_space(3))) v4i_vs*)(l), 0, 0);
#else
    *(v8bf*)l = *(const v8bf*)g;
#endif
}

static __device__ __forceinline__ void wait_async_copies() {
#if HAVE_ASYNC
#if HAVE_WAITASYNC
    __builtin_amdgcn_s_wait_asynccnt(0);
#else
    asm volatile("s_wait_asynccnt 0x0" ::: "memory");
#endif
#endif
}

static __device__ __forceinline__ v16bf combine16(v8bf lo, v8bf hi) {
    return __builtin_shufflevector(lo, hi, 0,1,2,3,4,5,6,7,8,9,10,11,12,13,14,15);
}

static __device__ __forceinline__ v8bf cvt8(float4 a, float4 b) {
    v8bf r;
    r[0]=(bf16)a.x; r[1]=(bf16)a.y; r[2]=(bf16)a.z; r[3]=(bf16)a.w;
    r[4]=(bf16)b.x; r[5]=(bf16)b.y; r[6]=(bf16)b.z; r[7]=(bf16)b.w;
    return r;
}

// A-fragment (16x32 bf16): lane l<16 -> row l, K {0..15}; l>=16 -> row l-16, K {16..31}
static __device__ __forceinline__ v16bf load_a_frag(const bf16* sA, int row_base, int lane) {
    int lr = lane & 15, kg = lane >> 4;
    const bf16* p = sA + (row_base + lr) * LDSTR + kg * 16;
    v8bf lo = *(const v8bf*)(p);        // K = kg*16 .. kg*16+7
    v8bf hi = *(const v8bf*)(p + 8);    // K = kg*16+8 .. kg*16+15
    return combine16(lo, hi);
}

// B-fragment (32x16 bf16): lane l -> col l%16, K = (l/16)*16 .. +15 (contiguous)
static __device__ __forceinline__ v16bf load_b_frag(const bf16* sB, int col_base, int lane) {
    int lr = lane & 15, kg = lane >> 4;
    const bf16* p = sB + (col_base + lr) * LDSTR + kg * 16;
    return combine16(*(const v8bf*)(p), *(const v8bf*)(p + 8));
}

// ---------------------------------------------------------------------------
// Init: zero output accumulator + per-expert token counters (ws/out poisoned)
// ---------------------------------------------------------------------------
__global__ __launch_bounds__(256)
void init_kernel(float* __restrict__ out, int n, int* __restrict__ counts) {
    int i = blockIdx.x * blockDim.x + threadIdx.x;
    if (i < NEXP) counts[i] = 0;
    for (; i < n; i += gridDim.x * blockDim.x) out[i] = 0.f;
}

// ---------------------------------------------------------------------------
// f32 -> bf16 conversion of x
// ---------------------------------------------------------------------------
__global__ __launch_bounds__(256)
void cvt_bf16_kernel(const float* __restrict__ in, bf16* __restrict__ out, int n) {
    for (int i = blockIdx.x * blockDim.x + threadIdx.x; i < n; i += gridDim.x * blockDim.x)
        out[i] = (bf16)in[i];
}

// ---------------------------------------------------------------------------
// Router: logits = x @ gate_W^T + b; top-2 softmax; append token to its two
// experts' compacted lists (index + combine weight). One wave per token.
// Slot order from atomics is run-varying but per-token results are invariant.
// ---------------------------------------------------------------------------
__global__ __launch_bounds__(256)
void moe_router(const float* __restrict__ x, const float* __restrict__ gW,
                const float* __restrict__ gb,
                int* __restrict__ idxlist,     // [NEXP, TOKENS]
                float* __restrict__ wlist,     // [NEXP, TOKENS]
                int* __restrict__ counts) {    // [NEXP]
    int gid  = blockIdx.x * blockDim.x + threadIdx.x;
    int t    = gid >> 5;
    int lane = gid & 31;
    if (t >= TOKENS) return;

    float acc[NEXP];
#pragma unroll
    for (int e = 0; e < NEXP; e++) acc[e] = 0.f;

    const float* xp = x + (size_t)t * DMODEL;
    for (int i = lane; i < DMODEL; i += 32) {
        float xv = xp[i];
#pragma unroll
        for (int e = 0; e < NEXP; e++) acc[e] += xv * gW[e * DMODEL + i];
    }
#pragma unroll
    for (int e = 0; e < NEXP; e++) {
#pragma unroll
        for (int off = 16; off > 0; off >>= 1)
            acc[e] += __shfl_down(acc[e], off, 32);
    }
    if (lane == 0) {
        float logit[NEXP];
#pragma unroll
        for (int e = 0; e < NEXP; e++) logit[e] = acc[e] + gb[e];
        int i0 = 0;
#pragma unroll
        for (int e = 1; e < NEXP; e++) if (logit[e] > logit[i0]) i0 = e;
        int i1 = -1;
#pragma unroll
        for (int e = 0; e < NEXP; e++)
            if (e != i0 && (i1 < 0 || logit[e] > logit[i1])) i1 = e;
        float m  = fmaxf(logit[i0], logit[i1]);
        float e0 = __expf(logit[i0] - m), e1 = __expf(logit[i1] - m);
        float inv = __builtin_amdgcn_rcpf(e0 + e1);
        int s0 = atomicAdd(&counts[i0], 1);
        idxlist[i0 * TOKENS + s0] = t;
        wlist[i0 * TOKENS + s0]   = e0 * inv;
        int s1 = atomicAdd(&counts[i1], 1);
        idxlist[i1 * TOKENS + s1] = t;
        wlist[i1 * TOKENS + s1]   = e1 * inv;
    }
}

// ---------------------------------------------------------------------------
// Stage A (per expert): H[slot] = silu(x[tok] @ Wg^T) * (x[tok] @ W1^T)
// over the expert's compacted token list. grid (TOKENS/BM, DFF/BN), 8 waves,
// each wave a 32x32 tile (2x2 WMMA). Blocks past the count exit early.
// ---------------------------------------------------------------------------
__global__ __launch_bounds__(256)
void moe_gate_up(const bf16* __restrict__ Xbf,
                 const float* __restrict__ Wg,   // [DFF, DMODEL] f32
                 const float* __restrict__ W1,   // [DFF, DMODEL] f32
                 bf16* __restrict__ H,           // [TOKENS, DFF] bf16 (compacted)
                 const int* __restrict__ idxlist,
                 const int* __restrict__ counts,
                 int expert) {
    const int cnt = counts[expert];
    const int m0  = blockIdx.x * BM;
    if (m0 >= cnt) return;                       // uniform early exit

    __shared__ bf16 sA[BM * LDSTR];
    __shared__ bf16 sG[BN * LDSTR];
    __shared__ bf16 sV[BN * LDSTR];

    const int tid  = threadIdx.x;
    const int wave = tid >> 5;
    const int lane = tid & 31;
    const int n0   = blockIdx.y * BN;
    const int wm   = wave >> 1;
    const int wn   = wave & 1;
    const int* idx = idxlist + (size_t)expert * TOKENS;

    // gather source row for this thread's X-tile row (fixed across K loop)
    const int arow  = tid >> 1;
    const int acol  = (tid & 1) * 16;
    const int aslot = m0 + arow;
    const int atok  = (aslot < cnt) ? idx[aslot] : 0;
    const bf16* asrc = Xbf + (size_t)atok * DMODEL + acol;
    bf16* adst = &sA[arow * LDSTR + acol];

    const int wrow = tid >> 2;
    const int wcol = (tid & 3) * 8;
    const float* gsrc = Wg + (size_t)(n0 + wrow) * DMODEL + wcol;
    const float* vsrc = W1 + (size_t)(n0 + wrow) * DMODEL + wcol;

    v8f accG[2][2], accV[2][2];
#pragma unroll
    for (int i = 0; i < 2; i++)
#pragma unroll
        for (int j = 0; j < 2; j++)
#pragma unroll
            for (int r = 0; r < 8; r++) { accG[i][j][r] = 0.f; accV[i][j][r] = 0.f; }

    for (int k0 = 0; k0 < DMODEL; k0 += BK) {
        __syncthreads();
        // X tile 128x32 bf16 (gathered rows): async DMA straight into LDS
        copy16_g2l(asrc + k0,     adst);
        copy16_g2l(asrc + k0 + 8, adst + 8);
        {   // Wg/W1 tiles 64x32 f32 -> bf16 (overlaps with async DMA above)
            const float* g = gsrc + k0;
            const float* v = vsrc + k0;
            *(v8bf*)(&sG[wrow * LDSTR + wcol]) = cvt8(*(const float4*)(g), *(const float4*)(g + 4));
            *(v8bf*)(&sV[wrow * LDSTR + wcol]) = cvt8(*(const float4*)(v), *(const float4*)(v + 4));
            if (k0 + BK < DMODEL) {
                __builtin_prefetch(g + BK, 0, 1);
                __builtin_prefetch(v + BK, 0, 1);
            }
        }
        wait_async_copies();
        __syncthreads();

        v16bf af[2], gf[2], vf[2];
#pragma unroll
        for (int im = 0; im < 2; im++) af[im] = load_a_frag(sA, wm * 32 + im * 16, lane);
#pragma unroll
        for (int in = 0; in < 2; in++) {
            gf[in] = load_b_frag(sG, wn * 32 + in * 16, lane);
            vf[in] = load_b_frag(sV, wn * 32 + in * 16, lane);
        }
#pragma unroll
        for (int im = 0; im < 2; im++)
#pragma unroll
            for (int in = 0; in < 2; in++) {
                accG[im][in] = __builtin_amdgcn_wmma_f32_16x16x32_bf16(
                    false, af[im], false, gf[in], (short)0, accG[im][in], false, false);
                accV[im][in] = __builtin_amdgcn_wmma_f32_16x16x32_bf16(
                    false, af[im], false, vf[in], (short)0, accV[im][in], false, false);
            }
    }

    // epilogue: h = silu(g)*v via v_rcp (avoid IEEE div chains)
    const int lr = lane & 15, lg = lane >> 4;
#pragma unroll
    for (int im = 0; im < 2; im++)
#pragma unroll
        for (int in = 0; in < 2; in++)
#pragma unroll
            for (int r = 0; r < 8; r++) {
                int slot = m0 + wm * 32 + im * 16 + lg * 8 + r;
                int ng   = n0 + wn * 32 + in * 16 + lr;
                float g = accG[im][in][r];
                float v = accV[im][in][r];
                float sig = __builtin_amdgcn_rcpf(1.f + __expf(-g));
                H[(size_t)slot * DFF + ng] = (bf16)(g * sig * v);
            }
}

// ---------------------------------------------------------------------------
// Stage B (per expert): out[tok] += (H[slot] @ W2^T) * w[slot]
// grid (TOKENS/BM, DMODEL/BN). Scatter-accumulate guarded by slot < count.
// ---------------------------------------------------------------------------
__global__ __launch_bounds__(256)
void moe_down(const bf16* __restrict__ H,      // [TOKENS, DFF] bf16 (compacted)
              const float* __restrict__ W2,    // [DMODEL, DFF] f32
              const int* __restrict__ idxlist,
              const float* __restrict__ wlist,
              const int* __restrict__ counts,
              float* __restrict__ out,         // [TOKENS, DMODEL] f32
              int expert) {
    const int cnt = counts[expert];
    const int m0  = blockIdx.x * BM;
    if (m0 >= cnt) return;

    __shared__ bf16 sA[BM * LDSTR];
    __shared__ bf16 sB[BN * LDSTR];

    const int tid  = threadIdx.x;
    const int wave = tid >> 5;
    const int lane = tid & 31;
    const int n0   = blockIdx.y * BN;
    const int wm   = wave >> 1;
    const int wn   = wave & 1;
    const int*   idx = idxlist + (size_t)expert * TOKENS;
    const float* wl  = wlist   + (size_t)expert * TOKENS;

    const int arow = tid >> 1;
    const int acol = (tid & 1) * 16;
    const bf16* asrc = H + (size_t)(m0 + arow) * DFF + acol;
    bf16* adst = &sA[arow * LDSTR + acol];

    const int wrow = tid >> 2;
    const int wcol = (tid & 3) * 8;
    const float* wsrc = W2 + (size_t)(n0 + wrow) * DFF + wcol;

    v8f acc[2][2];
#pragma unroll
    for (int i = 0; i < 2; i++)
#pragma unroll
        for (int j = 0; j < 2; j++)
#pragma unroll
            for (int r = 0; r < 8; r++) acc[i][j][r] = 0.f;

    for (int k0 = 0; k0 < DFF; k0 += BK) {
        __syncthreads();
        // H tile 128x32 bf16: async DMA straight into LDS
        copy16_g2l(asrc + k0,     adst);
        copy16_g2l(asrc + k0 + 8, adst + 8);
        {   // W2 tile 64x32 f32 -> bf16 (overlaps with async DMA above)
            const float* w = wsrc + k0;
            *(v8bf*)(&sB[wrow * LDSTR + wcol]) = cvt8(*(const float4*)(w), *(const float4*)(w + 4));
            if (k0 + BK < DFF) __builtin_prefetch(w + BK, 0, 1);
        }
        wait_async_copies();
        __syncthreads();

        v16bf af[2], bf_[2];
#pragma unroll
        for (int im = 0; im < 2; im++) af[im]  = load_a_frag(sA, wm * 32 + im * 16, lane);
#pragma unroll
        for (int in = 0; in < 2; in++) bf_[in] = load_b_frag(sB, wn * 32 + in * 16, lane);
#pragma unroll
        for (int im = 0; im < 2; im++)
#pragma unroll
            for (int in = 0; in < 2; in++)
                acc[im][in] = __builtin_amdgcn_wmma_f32_16x16x32_bf16(
                    false, af[im], false, bf_[in], (short)0, acc[im][in], false, false);
    }

    const int lr = lane & 15, lg = lane >> 4;
#pragma unroll
    for (int im = 0; im < 2; im++)
#pragma unroll
        for (int in = 0; in < 2; in++)
#pragma unroll
            for (int r = 0; r < 8; r++) {
                int slot = m0 + wm * 32 + im * 16 + lg * 8 + r;
                if (slot < cnt) {
                    int tok  = idx[slot];
                    float wv = wl[slot];
                    int ng   = n0 + wn * 32 + in * 16 + lr;
                    size_t o = (size_t)tok * DMODEL + ng;
                    out[o] += acc[im][in][r] * wv;   // serialized across expert launches
                }
            }
}

// ---------------------------------------------------------------------------
// Launch
// ---------------------------------------------------------------------------
extern "C" void kernel_launch(void* const* d_in, const int* in_sizes, int n_in,
                              void* d_out, int out_size, void* d_ws, size_t ws_size,
                              hipStream_t stream) {
    const float* x     = (const float*)d_in[0];
    const float* gateW = (const float*)d_in[1];
    const float* gateB = (const float*)d_in[2];
    const float* Wg    = (const float*)d_in[3];
    const float* W1    = (const float*)d_in[4];
    const float* W2    = (const float*)d_in[5];
    float* out = (float*)d_out;

    char* ws = (char*)d_ws;
    const size_t XBF_BYTES = (size_t)TOKENS * DMODEL * sizeof(bf16);       // 16 MB
    const size_t IDX_BYTES = (size_t)NEXP * TOKENS * sizeof(int);          // 256 KB
    const size_t WL_BYTES  = (size_t)NEXP * TOKENS * sizeof(float);        // 256 KB
    const size_t CNT_BYTES = 256;
    bf16*  Xbf    = (bf16*)ws;
    int*   idxl   = (int*)(ws + XBF_BYTES);
    float* wl     = (float*)(ws + XBF_BYTES + IDX_BYTES);
    int*   counts = (int*)(ws + XBF_BYTES + IDX_BYTES + WL_BYTES);
    bf16*  H      = (bf16*)(ws + XBF_BYTES + IDX_BYTES + WL_BYTES + CNT_BYTES); // 64 MB

    // 1) zero output accumulator + counters
    init_kernel<<<4096, 256, 0, stream>>>(out, TOKENS * DMODEL, counts);
    // 2) x -> bf16
    cvt_bf16_kernel<<<4096, 256, 0, stream>>>(x, Xbf, TOKENS * DMODEL);
    // 3) router: top-2 softmax + token compaction per expert
    moe_router<<<TOKENS / 8, 256, 0, stream>>>(x, gateW, gateB, idxl, wl, counts);
    // 4) per-expert SwiGLU GEMMs over compacted token lists (stream-serialized)
    for (int e = 0; e < NEXP; e++) {
        const float* Wge = Wg + (size_t)e * DFF * DMODEL;
        const float* W1e = W1 + (size_t)e * DFF * DMODEL;
        const float* W2e = W2 + (size_t)e * DMODEL * DFF;
        moe_gate_up<<<dim3(TOKENS / BM, DFF / BN), 256, 0, stream>>>(
            Xbf, Wge, W1e, H, idxl, counts, e);
        moe_down<<<dim3(TOKENS / BM, DMODEL / BN), 256, 0, stream>>>(
            H, W2e, idxl, wl, counts, out, e);
    }
}
